// MultiHeadAttentionNoTE_15831249453406
// MI455X (gfx1250) — matrix operand order repaired and checked by reading, not compiled, not run on hardware
//
#include <hip/hip_runtime.h>
#include <hip/hip_bf16.h>
#include <math.h>

// ---------------------------------------------------------------------------
// MHA forward for MI455X (gfx1250, wave32, WMMA 16x16x32 f16->f32)
// B=2, S=2048, DIM=2048, H=32, HD=64
// ---------------------------------------------------------------------------

typedef _Float16 v8h  __attribute__((ext_vector_type(8)));
typedef _Float16 v16h __attribute__((ext_vector_type(16)));
typedef float    v8f  __attribute__((ext_vector_type(8)));
typedef unsigned uint32x4 __attribute__((ext_vector_type(4)));
typedef int      int32x4_t __attribute__((ext_vector_type(4)));
typedef int      int32x8_t __attribute__((ext_vector_type(8)));

#if __has_builtin(__builtin_amdgcn_tensor_load_to_lds) && \
    __has_builtin(__builtin_amdgcn_s_wait_tensorcnt)
#define USE_TDM 1
#else
#define USE_TDM 0
#endif

static __device__ __forceinline__ v16h cat16(v8h a, v8h b) {
    return __builtin_shufflevector(a, b, 0,1,2,3,4,5,6,7,8,9,10,11,12,13,14,15);
}

static __device__ __forceinline__ v8f wmma_f16(v16h a, v16h b, v8f c) {
    return __builtin_amdgcn_wmma_f32_16x16x32_f16(false, a, false, b,
                                                  (short)0, c, false, false);
}

#if USE_TDM
// 2-D TDM load: tile (tile_w x tile_h) of f16 from row-major tensor
// (row stride = stride_elems) at gaddr into LDS at lds_byte_off, packed
// contiguously (row length tile_w). Descriptor per CDNA5 ISA §8.3-8.5.
static __device__ __forceinline__ void tdm_load_2d_f16(
    const void* gaddr, unsigned lds_byte_off,
    unsigned tile_w, unsigned tile_h,
    unsigned tensor_w, unsigned tensor_h, unsigned stride_elems) {
    unsigned long long ga = (unsigned long long)(size_t)gaddr;
    uint32x4 g0;
    g0[0] = 1u;                                   // count=1 (valid), user mode
    g0[1] = lds_byte_off;                         // lds_addr
    g0[2] = (unsigned)(ga & 0xffffffffu);         // global_addr[31:0]
    g0[3] = (unsigned)((ga >> 32) & 0x1ffffffu)   // global_addr[56:32]
          | (2u << 30);                           // type=2 ("image")
    int32x8_t g1;
    g1[0] = (int)(1u << 16);                      // data_size=1 (2B), mask=0
    g1[1] = (int)((tensor_w & 0xffffu) << 16);    // tensor_dim0[15:0]
    g1[2] = (int)(((tensor_w >> 16) & 0xffffu)    // tensor_dim0[31:16]
          | ((tensor_h & 0xffffu) << 16));        // tensor_dim1[15:0]
    g1[3] = (int)(((tensor_h >> 16) & 0xffffu)    // tensor_dim1[31:16]
          | ((tile_w & 0xffffu) << 16));          // tile_dim0
    g1[4] = (int)(tile_h & 0xffffu);              // tile_dim1 (tile_dim2=0)
    g1[5] = (int)stride_elems;                    // tensor_dim0_stride[31:0]
    g1[6] = 0;                                    // stride hi / dim1_stride lo
    g1[7] = 0;
    int32x4_t z4 = {0, 0, 0, 0};
#if defined(__clang_major__) && (__clang_major__ >= 23)
    int32x8_t z8 = {0, 0, 0, 0, 0, 0, 0, 0};
    __builtin_amdgcn_tensor_load_to_lds(g0, g1, z4, z4, z8, 0);
#else
    __builtin_amdgcn_tensor_load_to_lds(g0, g1, z4, z4, 0);
#endif
}
#endif

// ---------------------------------------------------------------------------
// fp32 -> fp16 cast
// ---------------------------------------------------------------------------
__global__ void cast_f2h_kernel(const float* __restrict__ in,
                                _Float16* __restrict__ out, long n) {
    long i = (long)blockIdx.x * blockDim.x + threadIdx.x;
    if (i < n) out[i] = (_Float16)in[i];
}

// ---------------------------------------------------------------------------
// C[M,N] = A[M,K] * B[N,K]^T   (A,B f16 row-major; C = f16 or f32)
// Block tile 128x128, BK=32. 256 threads = 8 waves in 4(M) x 2(N) grid;
// each wave owns a 32x64 strip = 2x4 WMMA accumulators.
// A tile staged by Tensor Data Mover (wave 0); B tile staged by all threads.
// ---------------------------------------------------------------------------
template <typename OutT>
__global__ __launch_bounds__(256)
void gemm_abT_kernel(const _Float16* __restrict__ A,
                     const _Float16* __restrict__ Bm,
                     OutT* __restrict__ C,
                     int M, int N, int K) {
    constexpr int BM = 128, BN = 128, BK = 32;
    constexpr int LDA = BK;       // unpadded: TDM writes tile contiguously
    constexpr int LDB = BK + 8;   // 40 halves = 80B (16B multiple)
    __shared__ __attribute__((aligned(128))) _Float16 As[BM * LDA];
    __shared__ __attribute__((aligned(32)))  _Float16 Bs[BN * LDB];

    const int tid  = threadIdx.x;
    const int lane = tid & 31;
    const int wave = tid >> 5;          // 0..7
    const int wm   = wave >> 1;         // 0..3  (M strip)
    const int wn   = wave & 1;          // 0..1  (N strip)
    const int lh   = lane >> 4;         // half of wave: 0/1
    const int lr   = lane & 15;
    const int mBase = blockIdx.y * BM;
    const int nBase = blockIdx.x * BN;

    // B staging: 128 rows x 32 halves, 16 halves per thread
    const int brow = tid >> 1, bc = (tid & 1) * 16;
#if !USE_TDM
    const int arow = tid >> 1, ac = (tid & 1) * 16;
#endif

    v8f acc[2][4] = {};

    for (int k0 = 0; k0 < K; k0 += BK) {
        __syncthreads();
#if USE_TDM
        if (wave == 0) {
            tdm_load_2d_f16(A + (size_t)mBase * K + k0,
                            (unsigned)(size_t)(void*)As,
                            BK, BM, (unsigned)K, (unsigned)M, (unsigned)K);
        }
#else
        {
            const _Float16* ga = A + (size_t)(mBase + arow) * K + k0 + ac;
            *(v8h*)&As[arow * LDA + ac]     = *(const v8h*)(ga);
            *(v8h*)&As[arow * LDA + ac + 8] = *(const v8h*)(ga + 8);
        }
#endif
        {
            const _Float16* gb = Bm + (size_t)(nBase + brow) * K + k0 + bc;
            *(v8h*)&Bs[brow * LDB + bc]     = *(const v8h*)(gb);
            *(v8h*)&Bs[brow * LDB + bc + 8] = *(const v8h*)(gb + 8);
            if (k0 + BK < K) __builtin_prefetch(gb + BK, 0, 1);
        }
#if USE_TDM
        if (wave == 0) __builtin_amdgcn_s_wait_tensorcnt(0);
#endif
        __syncthreads();

        // A fragments: lane lr = row within 16-row tile; elem j -> k=lh*16+j
        v16h afrag[2];
#pragma unroll
        for (int mi = 0; mi < 2; ++mi) {
            const _Float16* pa = &As[(wm * 32 + mi * 16 + lr) * LDA];
            afrag[mi] = cat16(*(const v8h*)(pa + lh * 16),
                              *(const v8h*)(pa + lh * 16 + 8));
        }
#pragma unroll
        for (int nt = 0; nt < 4; ++nt) {
            // B fragment: lane lr = col; elem j -> k = lh*16+j
            const _Float16* pb = &Bs[(wn * 64 + nt * 16 + lr) * LDB + lh * 16];
            v16h bfrag = cat16(*(const v8h*)pb, *(const v8h*)(pb + 8));
#pragma unroll
            for (int mi = 0; mi < 2; ++mi)
                acc[mi][nt] = wmma_f16(afrag[mi], bfrag, acc[mi][nt]);
        }
    }

    // C/D layout: VGPR i, lanes 0-15 -> M=i, lanes 16-31 -> M=8+i; N=lr
#pragma unroll
    for (int mi = 0; mi < 2; ++mi) {
#pragma unroll
        for (int nt = 0; nt < 4; ++nt) {
#pragma unroll
            for (int i = 0; i < 8; ++i) {
                int m = mBase + wm * 32 + mi * 16 + lh * 8 + i;
                int n = nBase + wn * 64 + nt * 16 + lr;
                C[(size_t)m * N + n] = (OutT)acc[mi][nt][i];
            }
        }
    }
}

// ---------------------------------------------------------------------------
// RoPE applied in-place to q and k slices of fused qkv [B,S,3*H*HD]
// ---------------------------------------------------------------------------
__global__ void rope_kernel(_Float16* __restrict__ qkv) {
    long idx = (long)blockIdx.x * blockDim.x + threadIdx.x;
    const long total = (long)2 * 2048 * 2 * 32 * 32;  // B*S*2*H*(HD/2)
    if (idx >= total) return;
    int i     = (int)(idx & 31); idx >>= 5;
    int h     = (int)(idx & 31); idx >>= 5;
    int which = (int)(idx & 1);  idx >>= 1;   // 0=q, 1=k
    int s     = (int)(idx & 2047); idx >>= 11;
    int b     = (int)idx;

    float inv = __powf(10000.0f, -(float)i / 32.0f);
    float sinv, cosv;
    __sincosf((float)s * inv, &sinv, &cosv);

    _Float16* base = qkv + ((size_t)(b * 2048 + s)) * 6144 + which * 2048 + h * 64;
    float x1 = (float)base[i];
    float x2 = (float)base[i + 32];
    base[i]      = (_Float16)(x1 * cosv - x2 * sinv);
    base[i + 32] = (_Float16)(x1 * sinv + x2 * cosv);
}

// ---------------------------------------------------------------------------
// Causal flash attention. Grid: (S/64, B*H). Block: 128 threads = 4 waves.
// Wave w owns q rows [qb*64 + w*16, +16). KV tiles of 32 rows.
// K staged row-major; V staged TRANSPOSED ([d][kv]) so the P*V B-fragment
// is two contiguous ds_load_b128s instead of a 16-way scalar gather.
// ---------------------------------------------------------------------------
__global__ __launch_bounds__(128)
void attn_kernel(const _Float16* __restrict__ qkv, _Float16* __restrict__ out) {
    constexpr int S = 2048, E3 = 6144;
    constexpr int KLD = 40;  // Kt row stride (halves): 80B, 16B multiple
    constexpr int VLD = 40;  // VtT row stride
    __shared__ __attribute__((aligned(32))) _Float16 Kt[32 * KLD];
    __shared__ __attribute__((aligned(32))) _Float16 VtT[64 * VLD];
    __shared__ __attribute__((aligned(32))) _Float16 Pb[4 * 16 * 32];

    const int tid  = threadIdx.x;
    const int lane = tid & 31;
    const int wave = tid >> 5;   // 0..3
    const int lh   = lane >> 4;
    const int lr   = lane & 15;
    const int qb   = blockIdx.x;        // 0..31
    const int bh   = blockIdx.y;        // 0..63
    const int b    = bh >> 5, h = bh & 31;
    const int qrow0 = qb * 64 + wave * 16;

    // Q fragments (A layout), loaded once: row = qrow0 + lr, d in [0,64)
    const _Float16* qp = qkv + (size_t)(b * S + qrow0 + lr) * E3 + h * 64;
    v16h aq0 = cat16(*(const v8h*)(qp + lh * 16),      *(const v8h*)(qp + lh * 16 + 8));
    v16h aq1 = cat16(*(const v8h*)(qp + 32 + lh * 16), *(const v8h*)(qp + 40 + lh * 16));

    v8f o[4] = {};
    float mrow[8], lrow[8];
#pragma unroll
    for (int i = 0; i < 8; ++i) { mrow[i] = -1e30f; lrow[i] = 0.0f; }

    const float SC = 0.125f;  // 1/sqrt(64)
    const int srow = tid >> 2, scol = (tid & 3) * 16;  // staging: 32 x 64 halves
    const int ntiles = qb * 2 + 2;

    for (int t = 0; t < ntiles; ++t) {
        const int kv0 = t * 32;
        __syncthreads();
        {
            const _Float16* gk =
                qkv + (size_t)(b * S + kv0 + srow) * E3 + 2048 + h * 64 + scol;
            const _Float16* gv =
                qkv + (size_t)(b * S + kv0 + srow) * E3 + 4096 + h * 64 + scol;
            *(v8h*)&Kt[srow * KLD + scol]     = *(const v8h*)gk;
            *(v8h*)&Kt[srow * KLD + scol + 8] = *(const v8h*)(gk + 8);
            v8h r0 = *(const v8h*)gv;
            v8h r1 = *(const v8h*)(gv + 8);
#pragma unroll
            for (int j = 0; j < 8; ++j) {
                VtT[(scol + j) * VLD + srow]     = r0[j];
                VtT[(scol + 8 + j) * VLD + srow] = r1[j];
            }
            if (t + 1 < ntiles) {
                __builtin_prefetch(gk + 32 * E3, 0, 1);
                __builtin_prefetch(gv + 32 * E3, 0, 1);
            }
        }
        __syncthreads();

        // scores: s0 cols kv0+[0,16), s1 cols kv0+[16,32); K-dim = HD = 64
        v8f s0 = {}, s1 = {};
        {
            const _Float16* pk0 = &Kt[lr * KLD + lh * 16];
            v16h b00 = cat16(*(const v8h*)pk0,        *(const v8h*)(pk0 + 8));
            v16h b01 = cat16(*(const v8h*)(pk0 + 32), *(const v8h*)(pk0 + 40));
            const _Float16* pk1 = &Kt[(16 + lr) * KLD + lh * 16];
            v16h b10 = cat16(*(const v8h*)pk1,        *(const v8h*)(pk1 + 8));
            v16h b11 = cat16(*(const v8h*)(pk1 + 32), *(const v8h*)(pk1 + 40));
            s0 = wmma_f16(aq0, b00, s0);
            s0 = wmma_f16(aq1, b01, s0);
            s1 = wmma_f16(aq0, b10, s1);
            s1 = wmma_f16(aq1, b11, s1);
        }

        // online softmax; row m = lh*8 + i lives in the 16-lane half, col = lr
#pragma unroll
        for (int i = 0; i < 8; ++i) {
            int qi = qrow0 + lh * 8 + i;
            float a0 = s0[i] * SC, a1 = s1[i] * SC;
            if (kv0 + lr > qi)      a0 = -1e30f;
            if (kv0 + 16 + lr > qi) a1 = -1e30f;
            float mx = fmaxf(a0, a1);
#pragma unroll
            for (int off = 1; off < 16; off <<= 1)
                mx = fmaxf(mx, __shfl_xor(mx, off, 32));
            float mnew = fmaxf(mrow[i], mx);
            float sc_old = __expf(mrow[i] - mnew);
            float e0 = __expf(a0 - mnew);
            float e1 = __expf(a1 - mnew);
            float rs = e0 + e1;
#pragma unroll
            for (int off = 1; off < 16; off <<= 1)
                rs += __shfl_xor(rs, off, 32);
            lrow[i] = lrow[i] * sc_old + rs;
            mrow[i] = mnew;
            o[0][i] *= sc_old; o[1][i] *= sc_old;
            o[2][i] *= sc_old; o[3][i] *= sc_old;
            _Float16* pp = &Pb[(wave * 16 + lh * 8 + i) * 32];
            pp[lr]      = (_Float16)e0;
            pp[16 + lr] = (_Float16)e1;
        }

        // P as WMMA A fragment (wave-private LDS region; in-wave DS ordering)
        const _Float16* ppr = &Pb[(wave * 16 + lr) * 32];
    v16h ap = cat16(*(const v8h*)(ppr + lh * 16),
                        *(const v8h*)(ppr + lh * 16 + 8));

        // o[nt] += P(16x32) x V(32x16): B elem j -> kv row lh*16+j, col nt*16+lr
#pragma unroll
        for (int nt = 0; nt < 4; ++nt) {
            const _Float16* pv = &VtT[(nt * 16 + lr) * VLD + lh * 16];
            v16h bv = cat16(*(const v8h*)pv, *(const v8h*)(pv + 8));
            o[nt] = wmma_f16(ap, bv, o[nt]);
        }
    }

    // normalize and store: out[b,s,h*64+d] (f16)
#pragma unroll
    for (int nt = 0; nt < 4; ++nt) {
#pragma unroll
        for (int i = 0; i < 8; ++i) {
            float val = o[nt][i] / lrow[i];
            int q = qrow0 + lh * 8 + i;
            out[(size_t)(b * S + q) * 2048 + h * 64 + nt * 16 + lr] = (_Float16)val;
        }
    }
}

// ---------------------------------------------------------------------------
// launcher
// ---------------------------------------------------------------------------
extern "C" void kernel_launch(void* const* d_in, const int* in_sizes, int n_in,
                              void* d_out, int out_size, void* d_ws, size_t ws_size,
                              hipStream_t stream) {
    const float* x     = (const float*)d_in[0];  // [2,2048,2048]
    const float* w_qkv = (const float*)d_in[1];  // [6144,2048]
    const float* w_o   = (const float*)d_in[2];  // [2048,2048]
    float* out = (float*)d_out;                  // [2,2048,2048]

    const long NX  = 8388608;    // 2*2048*2048
    const long NWQ = 12582912;   // 6144*2048
    const long NWO = 4194304;    // 2048*2048
    const long NQKV = 25165824;  // 4096*6144

    _Float16* x_h  = (_Float16*)d_ws;
    _Float16* wq_h = x_h + NX;
    _Float16* wo_h = wq_h + NWQ;
    _Float16* qkv  = wo_h + NWO;
    _Float16* ao   = qkv + NQKV;
    (void)n_in; (void)in_sizes; (void)out_size; (void)ws_size;

    cast_f2h_kernel<<<(NX + 255) / 256, 256, 0, stream>>>(x, x_h, NX);
    cast_f2h_kernel<<<(NWQ + 255) / 256, 256, 0, stream>>>(w_qkv, wq_h, NWQ);
    cast_f2h_kernel<<<(NWO + 255) / 256, 256, 0, stream>>>(w_o, wo_h, NWO);

    // qkv[b*S+s, e] = x . w_qkv^T : M=4096, N=6144, K=2048
    gemm_abT_kernel<_Float16><<<dim3(6144 / 128, 4096 / 128), 256, 0, stream>>>(
        x_h, wq_h, qkv, 4096, 6144, 2048);

    rope_kernel<<<(long)(2L * 2048 * 2 * 32 * 32 + 255) / 256, 256, 0, stream>>>(qkv);

    attn_kernel<<<dim3(2048 / 64, 2 * 32), 128, 0, stream>>>(qkv, ao);

    // out = attn_out . w_o^T : M=4096, N=2048, K=2048 (fp32 output)
    gemm_abT_kernel<float><<<dim3(2048 / 128, 4096 / 128), 256, 0, stream>>>(
        ao, wo_h, out, 4096, 2048, 2048);
}
